// SelfAttentionMemory_66967130079881
// MI455X (gfx1250) — matrix operand order repaired and checked by reading, not compiled, not run on hardware
//
#include <hip/hip_runtime.h>
#include <hip/hip_bf16.h>
#include <math.h>

// ---------------------------------------------------------------------------
// SA-ConvLSTM attention-memory cell for MI455X (gfx1250, wave32, WMMA).
// B=8, C=64, CI=32, H=W=48, N=2304.
// ---------------------------------------------------------------------------

typedef _Float16 v16h __attribute__((ext_vector_type(16)));
typedef _Float16 v8h  __attribute__((ext_vector_type(8)));
typedef float    v8f  __attribute__((ext_vector_type(8)));

#define N_SP   2304
#define BATCH  8
#define CDIM   64
#define CIDIM  32

// ---- fragment helpers (layouts per CDNA5 ISA 7.12.2, wave32) ---------------

__device__ inline v16h make_v16(v8h a, v8h b) {
    v16h r;
#pragma unroll
    for (int i = 0; i < 8; ++i) { r[i] = a[i]; r[i + 8] = b[i]; }
    return r;
}

// A-matrix 16x32 f16 from a row-major matrix (row stride = ldk halfs).
// lane L: row M = L%16; sel = L/16; elems 0..7 = K[k0+sel*8 .. +8),
// elems 8..15 = K[k0+16+sel*8 .. +8)  -> two aligned 16B loads.
__device__ inline v16h load_afrag_rowmajor(const _Float16* __restrict__ base,
                                           int ldk, int k0) {
    int lane = threadIdx.x & 31;
    int m = lane & 15, sel = lane >> 4;
    const _Float16* p = base + (size_t)m * ldk + k0 + sel * 8;
    v8h g0 = *(const v8h*)p;
    v8h g1 = *(const v8h*)(p + 16);
    return make_v16(g0, g1);
}

// Same A layout but from an LDS tile stored q-major [16][32].
__device__ inline v16h load_afrag_lds(const _Float16* t) {
    int lane = threadIdx.x & 31;
    int m = lane & 15, sel = lane >> 4;
    v8h g0 = *(const v8h*)(t + m * 32 + sel * 8);
    v8h g1 = *(const v8h*)(t + m * 32 + 16 + sel * 8);
    return make_v16(g0, g1);
}

// B-matrix 32x16 f16 from an LDS tile stored n-major [16][32] (n = column).
// lane L: column N = L%16; sel = L/16; elems 0..7 = K[sel*8 .. +8),
// elems 8..15 = K[16+sel*8 .. +8).
__device__ inline v16h load_bfrag_nmajor(const _Float16* t) {
    int lane = threadIdx.x & 31;
    int n = lane & 15, kb = (lane >> 4) * 8;
    const _Float16* p = t + n * 32 + kb;
    return make_v16(*(const v8h*)p, *(const v8h*)(p + 16));
}

// ---- f32 -> f16 conversion -------------------------------------------------

__global__ void cvt_kernel(const float* __restrict__ s, _Float16* __restrict__ d, int n) {
    for (int i = blockIdx.x * blockDim.x + threadIdx.x; i < n; i += gridDim.x * blockDim.x)
        d[i] = (_Float16)s[i];
}

// ---- generic WMMA GEMM: OUT[b] = W(MxK) @ X[b](KxN) + bias ----------------
// grid = (M/16, N/64, B), block = 128 (4 waves, one 16x16 tile each).

template <bool OUT_F16, bool ACCUM>
__global__ __launch_bounds__(128) void gemm_wmma(
    const _Float16* __restrict__ W, const _Float16* __restrict__ X,
    const float* __restrict__ bias, void* __restrict__ OUTv, int M, int K) {
    __shared__ _Float16 Xt[4][16 * 32];

    const int wave = threadIdx.x >> 5, lane = threadIdx.x & 31;
    const int mt = blockIdx.x, nt = blockIdx.y * 4 + wave, b = blockIdx.z;
    const int n0 = nt * 16;
    const _Float16* Xb = X + (size_t)b * K * N_SP;
    const _Float16* Wt = W + (size_t)mt * 16 * K;
    _Float16* t = Xt[wave];

    v8f acc = {};
    for (int k0 = 0; k0 < K; k0 += 32) {
        // stage X tile [32k x 16n] transposed into LDS as [16n][32k]
        const _Float16* src = Xb + (size_t)(k0 + lane) * N_SP + n0;
        v8h x0 = *(const v8h*)src;
        v8h x1 = *(const v8h*)(src + 8);
#pragma unroll
        for (int n = 0; n < 8; ++n) t[n * 32 + lane] = x0[n];
#pragma unroll
        for (int n = 0; n < 8; ++n) t[(n + 8) * 32 + lane] = x1[n];

        v16h a  = load_afrag_rowmajor(Wt, K, k0);
        v16h bx = load_bfrag_nmajor(t);
        acc = __builtin_amdgcn_wmma_f32_16x16x32_f16(false, a, false, bx,
                                                     (short)0, acc, false, false);
    }

    // C/D layout: lane L: col = n0 + L%16, rows = mt*16 + r + (L/16)*8
    const int col = n0 + (lane & 15);
    const int rowbase = mt * 16 + ((lane >> 4) << 3);
    if constexpr (OUT_F16) {
        _Float16* O = (_Float16*)OUTv + (size_t)b * M * N_SP;
#pragma unroll
        for (int r = 0; r < 8; ++r) {
            int row = rowbase + r;
            O[(size_t)row * N_SP + col] = (_Float16)(acc[r] + bias[row]);
        }
    } else {
        float* O = (float*)OUTv + (size_t)b * M * N_SP;
#pragma unroll
        for (int r = 0; r < 8; ++r) {
            int row = rowbase + r;
            size_t idx = (size_t)row * N_SP + col;
            float v = acc[r] + bias[row];
            if constexpr (ACCUM) v += O[idx];
            O[idx] = v;
        }
    }
}

// ---- fused flash attention:  Z = V @ softmax(Q^T K)^T ----------------------
// grid = (N/64, 2 /*h,m*/, B), block = 128.  Each wave owns 16 queries; the
// block cooperatively stages 32-key K/V panels in LDS.
// Online softmax uses a TILE-wide running max (valid upper bound; cancels in
// the normalization), making the rescale wave-uniform so row sums can be kept
// as per-lane partials and reduced once at the end.

__global__ __launch_bounds__(128) void attn_kernel(
    const _Float16* __restrict__ Q,
    const _Float16* __restrict__ Kh, const _Float16* __restrict__ Vh,
    const _Float16* __restrict__ Km, const _Float16* __restrict__ Vm,
    _Float16* __restrict__ Zcat) {
    __shared__ _Float16 Qt[4][16 * 32];   // per-wave Q^T tile, q-major
    __shared__ _Float16 Pt[4][16 * 32];   // per-wave P tile, q-major
    __shared__ _Float16 Kt[32 * 32];      // key-major [key][c]
    __shared__ _Float16 Vt[64 * 32];      // c-major   [c][key]

    const int tid = threadIdx.x;
    const int wave = tid >> 5, lane = tid & 31;
    const int b = blockIdx.z, sel = blockIdx.y;
    const int q0 = blockIdx.x * 64 + wave * 16;

    const _Float16* Qp = Q + (size_t)b * CIDIM * N_SP;
    const _Float16* Kp = (sel ? Km : Kh) + (size_t)b * CIDIM * N_SP;
    const _Float16* Vp = (sel ? Vm : Vh) + (size_t)b * CDIM * N_SP;

    // stage this wave's Q^T tile: Qt[q][c], c = lane
    {
        const _Float16* src = Qp + (size_t)lane * N_SP + q0;
        v8h a0 = *(const v8h*)src;
        v8h a1 = *(const v8h*)(src + 8);
        _Float16* qt = Qt[wave];
#pragma unroll
        for (int n = 0; n < 8; ++n) qt[n * 32 + lane] = a0[n];
#pragma unroll
        for (int n = 0; n < 8; ++n) qt[(n + 8) * 32 + lane] = a1[n];
    }
    const v16h aq = load_afrag_lds(Qt[wave]);

    v8f acc[4] = {v8f{}, v8f{}, v8f{}, v8f{}};
    float lp[8];                       // per-lane partial row sums
#pragma unroll
    for (int r = 0; r < 8; ++r) lp[r] = 0.0f;
    float mrun = -1e30f;               // wave-uniform running max

    _Float16* pt = Pt[wave];
    const int cl = lane & 15;

    for (int m0 = 0; m0 < N_SP; m0 += 32) {
        __syncthreads();
        // stage K panel transposed -> Kt[key][c]
        {
            int c = tid >> 2, kq = (tid & 3) * 8;
            v8h kv = *(const v8h*)(Kp + (size_t)c * N_SP + m0 + kq);
#pragma unroll
            for (int i = 0; i < 8; ++i) Kt[(kq + i) * 32 + c] = kv[i];
        }
        // stage V panel direct -> Vt[c][key]
#pragma unroll
        for (int i = 0; i < 2; ++i) {
            int idx = tid + i * 128;
            int c = idx >> 2, kq = (idx & 3) * 8;
            v8h vv = *(const v8h*)(Vp + (size_t)c * N_SP + m0 + kq);
            *(v8h*)(Vt + c * 32 + kq) = vv;
        }
        // prefetch next panel into cache while this one is consumed
        if (m0 + 32 < N_SP) {
            int c = tid >> 2, kq = (tid & 3) * 8;
            __builtin_prefetch(Kp + (size_t)c * N_SP + m0 + 32 + kq, 0, 0);
            __builtin_prefetch(Vp + (size_t)c * N_SP + m0 + 32 + kq, 0, 0);
            __builtin_prefetch(Vp + (size_t)(c + 32) * N_SP + m0 + 32 + kq, 0, 0);
        }
        __syncthreads();

        // two 16-key score tiles:  S = Q^T K   (K-dim = CI = 32, one WMMA each)
        v16h bk0 = load_bfrag_nmajor(Kt);
        v16h bk1 = load_bfrag_nmajor(Kt + 16 * 32);
        v8f z0 = {}, z1 = {};
        v8f s0 = __builtin_amdgcn_wmma_f32_16x16x32_f16(false, aq, false, bk0,
                                                        (short)0, z0, false, false);
        v8f s1 = __builtin_amdgcn_wmma_f32_16x16x32_f16(false, aq, false, bk1,
                                                        (short)0, z1, false, false);

        // tile-wide max: per-lane over 16 values, then 5-step butterfly (w=32)
        float rm = fmaxf(s0[0], s1[0]);
#pragma unroll
        for (int r = 1; r < 8; ++r) rm = fmaxf(rm, fmaxf(s0[r], s1[r]));
#pragma unroll
        for (int off = 16; off; off >>= 1) rm = fmaxf(rm, __shfl_xor(rm, off, 32));
        float nm = fmaxf(mrun, rm);
        float sc = __expf(mrun - nm);    // wave-uniform rescale
        mrun = nm;

#pragma unroll
        for (int r = 0; r < 8; ++r) {
            float p0 = __expf(s0[r] - nm);
            float p1 = __expf(s1[r] - nm);
            lp[r] = lp[r] * sc + p0 + p1;
            int q = r + ((lane >> 4) << 3);
            pt[q * 32 + cl]      = (_Float16)p0;
            pt[q * 32 + 16 + cl] = (_Float16)p1;
        }
#pragma unroll
        for (int cc = 0; cc < 4; ++cc) {
#pragma unroll
            for (int r = 0; r < 8; ++r) acc[cc][r] *= sc;
        }

        // Z += P @ V^T  (4 channel chunks of 16)
        v16h ap = load_afrag_lds(pt);
#pragma unroll
        for (int cc = 0; cc < 4; ++cc) {
            v16h bv = load_bfrag_nmajor(Vt + cc * 16 * 32);
            acc[cc] = __builtin_amdgcn_wmma_f32_16x16x32_f16(false, ap, false, bv,
                                                             (short)0, acc[cc],
                                                             false, false);
        }
    }

    // final row-sum reduction (once): rows split across lane halves -> w=16
    float rl[8];
#pragma unroll
    for (int r = 0; r < 8; ++r) {
        float s = lp[r];
#pragma unroll
        for (int off = 8; off; off >>= 1) s += __shfl_xor(s, off, 16);
        rl[r] = 1.0f / s;
    }

    // epilogue: scale by 1/l, store transposed into Zcat[b][sel*64 + c][n]
    _Float16* Zb = Zcat + ((size_t)b * 2 * CDIM + sel * CDIM) * N_SP;
    const int qb = ((lane >> 4) << 3);
#pragma unroll
    for (int cc = 0; cc < 4; ++cc) {
#pragma unroll
        for (int r = 0; r < 8; ++r) {
            Zb[(size_t)(cc * 16 + cl) * N_SP + q0 + qb + r] =
                (_Float16)(acc[cc][r] * rl[r]);
        }
    }
}

// ---- gate elementwise ------------------------------------------------------

__global__ void gates_kernel(const float* __restrict__ ip, const float* __restrict__ gp,
                             const float* __restrict__ op, const float* __restrict__ mprev,
                             float* __restrict__ Ht, float* __restrict__ Mt, int n) {
    int i = blockIdx.x * blockDim.x + threadIdx.x;
    if (i >= n) return;
    float it = 1.0f / (1.0f + __expf(-ip[i]));
    float gt = tanhf(gp[i]);
    float m = (1.0f - it) * mprev[i] + it * gt;
    float ot = 1.0f / (1.0f + __expf(-op[i]));
    Ht[i] = ot * m;
    Mt[i] = m;
}

// ---------------------------------------------------------------------------

extern "C" void kernel_launch(void* const* d_in, const int* in_sizes, int n_in,
                              void* d_out, int out_size, void* d_ws, size_t ws_size,
                              hipStream_t stream) {
    const float* ht = (const float*)d_in[0];
    const float* mt = (const float*)d_in[1];
    const float* W_q   = (const float*)d_in[2];  const float* b_q   = (const float*)d_in[3];
    const float* W_hk  = (const float*)d_in[4];  const float* b_hk  = (const float*)d_in[5];
    const float* W_hv  = (const float*)d_in[6];  const float* b_hv  = (const float*)d_in[7];
    const float* W_mk  = (const float*)d_in[8];  const float* b_mk  = (const float*)d_in[9];
    const float* W_mv  = (const float*)d_in[10]; const float* b_mv  = (const float*)d_in[11];
    const float* W_z   = (const float*)d_in[12]; const float* b_z   = (const float*)d_in[13];
    const float* W_mzo = (const float*)d_in[14]; const float* b_mzo = (const float*)d_in[15];
    const float* W_mho = (const float*)d_in[16]; const float* b_mho = (const float*)d_in[17];
    const float* W_mzg = (const float*)d_in[18]; const float* b_mzg = (const float*)d_in[19];
    const float* W_mhg = (const float*)d_in[20]; const float* b_mhg = (const float*)d_in[21];
    const float* W_mzi = (const float*)d_in[22]; const float* b_mzi = (const float*)d_in[23];
    const float* W_mhi = (const float*)d_in[24]; const float* b_mhi = (const float*)d_in[25];

    const size_t BCN  = (size_t)BATCH * CDIM * N_SP;   // 1179648
    const size_t BCIN = (size_t)BATCH * CIDIM * N_SP;  //  589824

    char* p = (char*)d_ws;
    auto alloc = [&](size_t bytes) -> void* {
        void* r = (void*)p;
        p += (bytes + 255) & ~(size_t)255;
        return r;
    };
    _Float16* ht16   = (_Float16*)alloc(BCN * 2);
    _Float16* mt16   = (_Float16*)alloc(BCN * 2);
    _Float16* q16    = (_Float16*)alloc(BCIN * 2);
    _Float16* kh16   = (_Float16*)alloc(BCIN * 2);
    _Float16* km16   = (_Float16*)alloc(BCIN * 2);
    _Float16* vh16   = (_Float16*)alloc(BCN * 2);
    _Float16* vm16   = (_Float16*)alloc(BCN * 2);
    _Float16* zcat16 = (_Float16*)alloc(2 * BCN * 2);
    _Float16* z16    = (_Float16*)alloc(BCN * 2);
    float* ipre = (float*)alloc(BCN * 4);
    float* gpre = (float*)alloc(BCN * 4);
    float* opre = (float*)alloc(BCN * 4);
    _Float16* wq16   = (_Float16*)alloc(32 * 64 * 2);
    _Float16* whk16  = (_Float16*)alloc(32 * 64 * 2);
    _Float16* whv16  = (_Float16*)alloc(64 * 64 * 2);
    _Float16* wmk16  = (_Float16*)alloc(32 * 64 * 2);
    _Float16* wmv16  = (_Float16*)alloc(64 * 64 * 2);
    _Float16* wz16   = (_Float16*)alloc(64 * 128 * 2);
    _Float16* wmzo16 = (_Float16*)alloc(64 * 64 * 2);
    _Float16* wmho16 = (_Float16*)alloc(64 * 64 * 2);
    _Float16* wmzg16 = (_Float16*)alloc(64 * 64 * 2);
    _Float16* wmhg16 = (_Float16*)alloc(64 * 64 * 2);
    _Float16* wmzi16 = (_Float16*)alloc(64 * 64 * 2);
    _Float16* wmhi16 = (_Float16*)alloc(64 * 64 * 2);

    auto cvt = [&](const float* s, _Float16* d, int n) {
        int blocks = (n + 255) / 256;
        if (blocks > 2048) blocks = 2048;
        cvt_kernel<<<blocks, 256, 0, stream>>>(s, d, n);
    };
    cvt(ht, ht16, (int)BCN);
    cvt(mt, mt16, (int)BCN);
    cvt(W_q, wq16, 32 * 64);     cvt(W_hk, whk16, 32 * 64);
    cvt(W_hv, whv16, 64 * 64);   cvt(W_mk, wmk16, 32 * 64);
    cvt(W_mv, wmv16, 64 * 64);   cvt(W_z, wz16, 64 * 128);
    cvt(W_mzo, wmzo16, 64 * 64); cvt(W_mho, wmho16, 64 * 64);
    cvt(W_mzg, wmzg16, 64 * 64); cvt(W_mhg, wmhg16, 64 * 64);
    cvt(W_mzi, wmzi16, 64 * 64); cvt(W_mhi, wmhi16, 64 * 64);

    const dim3 blk(128);
    const dim3 g32(32 / 16, N_SP / 64, BATCH);
    const dim3 g64(64 / 16, N_SP / 64, BATCH);

    // projections (f16 out, +bias)
    gemm_wmma<true, false><<<g32, blk, 0, stream>>>(wq16,  ht16, b_q,  q16,  32, 64);
    gemm_wmma<true, false><<<g32, blk, 0, stream>>>(whk16, ht16, b_hk, kh16, 32, 64);
    gemm_wmma<true, false><<<g64, blk, 0, stream>>>(whv16, ht16, b_hv, vh16, 64, 64);
    gemm_wmma<true, false><<<g32, blk, 0, stream>>>(wmk16, mt16, b_mk, km16, 32, 64);
    gemm_wmma<true, false><<<g64, blk, 0, stream>>>(wmv16, mt16, b_mv, vm16, 64, 64);

    // fused dual flash attention -> Zcat (f16, [B][128][N])
    attn_kernel<<<dim3(N_SP / 64, 2, BATCH), blk, 0, stream>>>(q16, kh16, vh16,
                                                               km16, vm16, zcat16);

    // Z = Wz @ Zcat + bz (K = 128)
    gemm_wmma<true, false><<<g64, blk, 0, stream>>>(wz16, zcat16, b_z, z16, 64, 128);

    // gate pre-activations: *_pre = Wz*@Z + b  then  += Wh*@ht + b
    gemm_wmma<false, false><<<g64, blk, 0, stream>>>(wmzi16, z16,  b_mzi, ipre, 64, 64);
    gemm_wmma<false, true ><<<g64, blk, 0, stream>>>(wmhi16, ht16, b_mhi, ipre, 64, 64);
    gemm_wmma<false, false><<<g64, blk, 0, stream>>>(wmzg16, z16,  b_mzg, gpre, 64, 64);
    gemm_wmma<false, true ><<<g64, blk, 0, stream>>>(wmhg16, ht16, b_mhg, gpre, 64, 64);
    gemm_wmma<false, false><<<g64, blk, 0, stream>>>(wmzo16, z16,  b_mzo, opre, 64, 64);
    gemm_wmma<false, true ><<<g64, blk, 0, stream>>>(wmho16, ht16, b_mho, opre, 64, 64);

    // gates + state update -> d_out = [Ht | Mt]
    float* Ht = (float*)d_out;
    float* Mt = Ht + BCN;
    gates_kernel<<<(int)((BCN + 255) / 256), 256, 0, stream>>>(ipre, gpre, opre,
                                                               mt, Ht, Mt, (int)BCN);
}
